// ExpertParallelMoE_5171140624830
// MI455X (gfx1250) — matrix operand order repaired and checked by reading, not compiled, not run on hardware
//
#include <hip/hip_runtime.h>
#include <hip/hip_bf16.h>

// ---------------- problem constants ----------------
constexpr int Dd = 1024;   // model dim
constexpr int Ff = 4096;   // ffn dim = 4*D
constexpr int Tt = 4096;   // tokens = B*S
constexpr int Ee = 8;      // experts
// GEMM tiling
constexpr int BM = 64;     // block tile M (tokens)
constexpr int BN = 256;    // block tile N
constexpr int BK = 64;     // staged K depth (two WMMA K-substeps)
constexpr int PITCH = 72;  // LDS row pitch in bf16 elems (144B, 16B-aligned chunks)

// ---------------- workspace layout (bytes) ----------------
constexpr size_t OFF_COUNTS  = 0;                               // E ints
constexpr size_t OFF_OFFSETS = 64;                              // E+1 ints
constexpr size_t OFF_TOK     = 128;                             // E*T ints
constexpr size_t OFF_GATE    = OFF_TOK  + (size_t)Ee*Tt*4;      // E*T floats
constexpr size_t OFF_W1B     = ((OFF_GATE + (size_t)Ee*Tt*4) + 255) & ~size_t(255); // E*F*D bf16 (N-major)
constexpr size_t OFF_W2B     = OFF_W1B + (size_t)Ee*Dd*Ff*2;    // E*D*F bf16 (N-major)
constexpr size_t OFF_H       = OFF_W2B + (size_t)Ee*Dd*Ff*2;    // 2*T*F bf16

// ---------------- types ----------------
typedef __attribute__((ext_vector_type(16))) __bf16 v16bf;
typedef __attribute__((ext_vector_type(8)))  float  v8f;
typedef __attribute__((ext_vector_type(8)))  unsigned short u16x8;

__device__ __forceinline__ unsigned short f2b(float f) {
    union { float f; unsigned u; } c; c.f = f;
    unsigned u = c.u;
    unsigned r = u + 0x7FFFu + ((u >> 16) & 1u);   // round-to-nearest-even
    return (unsigned short)(r >> 16);
}
__device__ __forceinline__ unsigned pack2(float a, float b) {
    return (unsigned)f2b(a) | ((unsigned)f2b(b) << 16);
}
__device__ __forceinline__ v16bf loadFrag(const unsigned short* p0, const unsigned short* p1) {
    union { u16x8 h[2]; v16bf v; } u;
    u.h[0] = *reinterpret_cast<const u16x8*>(p0);
    u.h[1] = *reinterpret_cast<const u16x8*>(p1);
    return u.v;
}

// ---------------- router: gates = softmax(x@Wg+bg), top-2, compact per expert ----------------
__global__ __launch_bounds__(256) void router_kernel(
    const float* __restrict__ x, const float* __restrict__ Wg, const float* __restrict__ bg,
    int* __restrict__ counts, int* __restrict__ tokList, float* __restrict__ gateList)
{
    const int w = threadIdx.x >> 5, lane = threadIdx.x & 31;
    const int t = blockIdx.x * 8 + w;
    float a[8] = {0,0,0,0,0,0,0,0};
    for (int d = lane; d < Dd; d += 32) {
        float xv = x[(size_t)t * Dd + d];
        const float4* wg = reinterpret_cast<const float4*>(Wg + (size_t)d * Ee);
        float4 w0 = wg[0], w1 = wg[1];
        a[0] += xv * w0.x; a[1] += xv * w0.y; a[2] += xv * w0.z; a[3] += xv * w0.w;
        a[4] += xv * w1.x; a[5] += xv * w1.y; a[6] += xv * w1.z; a[7] += xv * w1.w;
    }
    #pragma unroll
    for (int off = 16; off > 0; off >>= 1)
        #pragma unroll
        for (int e = 0; e < 8; e++) a[e] += __shfl_xor(a[e], off, 32);
    if (lane == 0) {
        float p[8], mx = -1e30f, s = 0.f;
        #pragma unroll
        for (int e = 0; e < 8; e++) { p[e] = a[e] + bg[e]; mx = p[e] > mx ? p[e] : mx; }
        #pragma unroll
        for (int e = 0; e < 8; e++) { p[e] = __expf(p[e] - mx); s += p[e]; }
        #pragma unroll
        for (int e = 0; e < 8; e++) p[e] /= s;
        int i1 = 0;
        #pragma unroll
        for (int e = 1; e < 8; e++) if (p[e] > p[i1]) i1 = e;
        int i2 = (i1 == 0) ? 1 : 0;
        #pragma unroll
        for (int e = 0; e < 8; e++) if (e != i1 && p[e] > p[i2]) i2 = e;
        int pos = atomicAdd(&counts[i1], 1);
        tokList[i1 * Tt + pos] = t; gateList[i1 * Tt + pos] = p[i1];
        pos = atomicAdd(&counts[i2], 1);
        tokList[i2 * Tt + pos] = t; gateList[i2 * Tt + pos] = p[i2];
    }
}

__global__ void prefix_kernel(const int* __restrict__ counts, int* __restrict__ offsets) {
    if (threadIdx.x == 0) {
        int s = 0;
        for (int e = 0; e < Ee; e++) { offsets[e] = s; s += counts[e]; }
        offsets[Ee] = s;
    }
}

// ---------------- fp32 [E][R][C] -> bf16 [E][C][R] transpose + convert (tiled) ----------------
__global__ __launch_bounds__(256) void transpose_cvt_kernel(
    const float* __restrict__ src, unsigned short* __restrict__ dst, int R, int C)
{
    const int e = blockIdx.z;
    __shared__ float tile[32][33];
    const int c0 = blockIdx.x * 32, r0 = blockIdx.y * 32;
    const float* s = src + (size_t)e * R * C;
    unsigned short* d = dst + (size_t)e * R * C;
    #pragma unroll
    for (int i = 0; i < 4; i++) {
        int r = r0 + threadIdx.y + i * 8;
        tile[threadIdx.y + i * 8][threadIdx.x] = s[(size_t)r * C + c0 + threadIdx.x];
    }
    __syncthreads();
    #pragma unroll
    for (int i = 0; i < 4; i++) {
        int c = c0 + threadIdx.y + i * 8;
        d[(size_t)c * R + r0 + threadIdx.x] = f2b(tile[threadIdx.x][threadIdx.y + i * 8]);
    }
}

// ---------------- shared WMMA inner step: 32x64 wave tile over one BK stage ----------------
__device__ __forceinline__ void mma_stage(
    const unsigned short* __restrict__ Al, const unsigned short* __restrict__ Bl,
    int waveM, int waveN, int half, int lr, v8f acc[2][4])
{
    #pragma unroll
    for (int kk = 0; kk < 2; kk++) {           // two K=32 substeps within BK=64
        const int koff = kk * 32;
        v16bf aF[2];
        #pragma unroll
        for (int mi = 0; mi < 2; mi++) {
            const unsigned short* aRow = &Al[(waveM * 32 + mi * 16 + lr) * PITCH + koff];
            aF[mi] = loadFrag(aRow + half * 8, aRow + 16 + half * 8);
        }
        #pragma unroll
        for (int j = 0; j < 4; j++) {
            const unsigned short* bRow = &Bl[(waveN * 64 + j * 16 + lr) * PITCH + koff + half * 8];
            v16bf bF = loadFrag(bRow, bRow + 16);
            acc[0][j] = __builtin_amdgcn_wmma_f32_16x16x32_bf16(
                false, aF[0], false, bF, (short)0, acc[0][j], false, false);
            acc[1][j] = __builtin_amdgcn_wmma_f32_16x16x32_bf16(
                false, aF[1], false, bF, (short)0, acc[1][j], false, false);
        }
    }
}

// ---------------- pass 1: h = relu(Xe @ W1e + b1e), bf16 out ----------------
__global__ __launch_bounds__(256) void gemm1_kernel(
    const float* __restrict__ x, const unsigned short* __restrict__ W1b,
    const float* __restrict__ b1, const int* __restrict__ counts, const int* __restrict__ offsets,
    const int* __restrict__ tokList, unsigned short* __restrict__ h)
{
    const int e = blockIdx.z;
    const int n_e = counts[e];
    const int m0 = blockIdx.x * BM;
    if (m0 >= n_e) return;
    const int n0 = blockIdx.y * BN;
    const int rowBase = offsets[e];

    __shared__ __align__(16) unsigned short Al[BM * PITCH];
    __shared__ __align__(16) unsigned short Bl[BN * PITCH];
    __shared__ int tokS[BM];

    const int tid = threadIdx.x;
    if (tid < BM) {
        int idx = m0 + tid; if (idx > n_e - 1) idx = n_e - 1;
        tokS[tid] = tokList[e * Tt + idx];
    }
    __syncthreads();

    const int lane = tid & 31, w = tid >> 5;
    const int waveM = w & 1, waveN = w >> 1;
    const int half = lane >> 4, lr = lane & 15;

    v8f acc[2][4];
    #pragma unroll
    for (int mi = 0; mi < 2; mi++)
        #pragma unroll
        for (int j = 0; j < 4; j++) acc[mi][j] = (v8f){0.f,0.f,0.f,0.f,0.f,0.f,0.f,0.f};

    const int am = tid >> 2;          // 0..63 A-row
    const int ac = (tid & 3) * 16;    // 0..48 A-col (16 elems per thread)
    const float* aSrc = x + (size_t)tokS[am] * Dd;
    const unsigned short* Wsrc = W1b + (size_t)e * Ff * Dd;

    for (int k0 = 0; k0 < Dd; k0 += BK) {
        const float4* af = reinterpret_cast<const float4*>(aSrc + k0 + ac);
        float4 f0 = af[0], f1 = af[1], f2 = af[2], f3 = af[3];
        __syncthreads();
        uint4 u0 = { pack2(f0.x,f0.y), pack2(f0.z,f0.w), pack2(f1.x,f1.y), pack2(f1.z,f1.w) };
        uint4 u1 = { pack2(f2.x,f2.y), pack2(f2.z,f2.w), pack2(f3.x,f3.y), pack2(f3.z,f3.w) };
        *reinterpret_cast<uint4*>(&Al[am * PITCH + ac + 0]) = u0;
        *reinterpret_cast<uint4*>(&Al[am * PITCH + ac + 8]) = u1;
        #pragma unroll
        for (int it = 0; it < 8; it++) {
            int idx2 = tid + 256 * it;
            int bn = idx2 >> 3, cc = idx2 & 7;
            uint4 bv = *reinterpret_cast<const uint4*>(Wsrc + (size_t)(n0 + bn) * Dd + k0 + cc * 8);
            *reinterpret_cast<uint4*>(&Bl[bn * PITCH + cc * 8]) = bv;
        }
        __syncthreads();
        if (k0 + BK < Dd)   // pull next B stage toward L0/L2 while we compute
            __builtin_prefetch(Wsrc + (size_t)(n0 + tid) * Dd + k0 + BK, 0, 1);
        mma_stage(Al, Bl, waveM, waveN, half, lr, acc);
    }
    #pragma unroll
    for (int mi = 0; mi < 2; mi++)
        #pragma unroll
        for (int j = 0; j < 4; j++) {
            int n = n0 + waveN * 64 + j * 16 + lr;
            float bias = b1[e * Ff + n];
            #pragma unroll
            for (int r = 0; r < 8; r++) {
                int rowLocal = waveM * 32 + mi * 16 + r + 8 * half;
                int m = m0 + rowLocal;
                if (m < n_e) {
                    float v = acc[mi][j][r] + bias;
                    v = v > 0.f ? v : 0.f;
                    h[(size_t)(rowBase + m) * Ff + n] = f2b(v);
                }
            }
        }
}

// ---------------- pass 2: out[t] += gate * (h @ W2e + b2e) ----------------
__global__ __launch_bounds__(256) void gemm2_kernel(
    const unsigned short* __restrict__ h, const unsigned short* __restrict__ W2b,
    const float* __restrict__ b2, const int* __restrict__ counts, const int* __restrict__ offsets,
    const int* __restrict__ tokList, const float* __restrict__ gateList, float* __restrict__ out)
{
    const int e = blockIdx.z;
    const int n_e = counts[e];
    const int m0 = blockIdx.x * BM;
    if (m0 >= n_e) return;
    const int n0 = blockIdx.y * BN;
    const int rowBase = offsets[e];

    __shared__ __align__(16) unsigned short Al[BM * PITCH];
    __shared__ __align__(16) unsigned short Bl[BN * PITCH];
    __shared__ int tokS[BM];
    __shared__ float gateS[BM];

    const int tid = threadIdx.x;
    if (tid < BM) {
        int idx = m0 + tid; if (idx > n_e - 1) idx = n_e - 1;
        tokS[tid]  = tokList[e * Tt + idx];
        gateS[tid] = gateList[e * Tt + idx];
    }
    __syncthreads();

    const int lane = tid & 31, w = tid >> 5;
    const int waveM = w & 1, waveN = w >> 1;
    const int half = lane >> 4, lr = lane & 15;

    v8f acc[2][4];
    #pragma unroll
    for (int mi = 0; mi < 2; mi++)
        #pragma unroll
        for (int j = 0; j < 4; j++) acc[mi][j] = (v8f){0.f,0.f,0.f,0.f,0.f,0.f,0.f,0.f};

    const int am = tid >> 2;          // 0..63
    const int ac = (tid & 3) * 16;    // 0..48
    int amRow = m0 + am; if (amRow > n_e - 1) amRow = n_e - 1;
    const unsigned short* aSrc = h + (size_t)(rowBase + amRow) * Ff;
    const unsigned short* Wsrc = W2b + (size_t)e * Dd * Ff;

    for (int k0 = 0; k0 < Ff; k0 += BK) {
        const uint4* af = reinterpret_cast<const uint4*>(aSrc + k0 + ac);
        uint4 a0 = af[0], a1 = af[1];
        __syncthreads();
        *reinterpret_cast<uint4*>(&Al[am * PITCH + ac + 0]) = a0;
        *reinterpret_cast<uint4*>(&Al[am * PITCH + ac + 8]) = a1;
        #pragma unroll
        for (int it = 0; it < 8; it++) {
            int idx2 = tid + 256 * it;
            int bn = idx2 >> 3, cc = idx2 & 7;
            uint4 bv = *reinterpret_cast<const uint4*>(Wsrc + (size_t)(n0 + bn) * Ff + k0 + cc * 8);
            *reinterpret_cast<uint4*>(&Bl[bn * PITCH + cc * 8]) = bv;
        }
        __syncthreads();
        if (k0 + BK < Ff)
            __builtin_prefetch(Wsrc + (size_t)(n0 + tid) * Ff + k0 + BK, 0, 1);
        mma_stage(Al, Bl, waveM, waveN, half, lr, acc);
    }
    #pragma unroll
    for (int mi = 0; mi < 2; mi++)
        #pragma unroll
        for (int j = 0; j < 4; j++) {
            int n = n0 + waveN * 64 + j * 16 + lr;
            float bias = b2[e * Dd + n];
            #pragma unroll
            for (int r = 0; r < 8; r++) {
                int rowLocal = waveM * 32 + mi * 16 + r + 8 * half;
                int m = m0 + rowLocal;
                if (m < n_e) {
                    float v = gateS[rowLocal] * (acc[mi][j][r] + bias);
                    atomicAdd(&out[(size_t)tokS[rowLocal] * Dd + n], v);
                }
            }
        }
}

// ---------------- launcher ----------------
extern "C" void kernel_launch(void* const* d_in, const int* in_sizes, int n_in,
                              void* d_out, int out_size, void* d_ws, size_t ws_size,
                              hipStream_t stream) {
    (void)in_sizes; (void)n_in; (void)ws_size;
    const float* x  = (const float*)d_in[0];
    const float* Wg = (const float*)d_in[1];
    const float* bg = (const float*)d_in[2];
    const float* W1 = (const float*)d_in[3];
    const float* b1 = (const float*)d_in[4];
    const float* W2 = (const float*)d_in[5];
    const float* b2 = (const float*)d_in[6];
    float* out = (float*)d_out;

    char* ws = (char*)d_ws;
    int*   counts   = (int*)(ws + OFF_COUNTS);
    int*   offsets  = (int*)(ws + OFF_OFFSETS);
    int*   tokList  = (int*)(ws + OFF_TOK);
    float* gateList = (float*)(ws + OFF_GATE);
    unsigned short* W1b  = (unsigned short*)(ws + OFF_W1B);
    unsigned short* W2b  = (unsigned short*)(ws + OFF_W2B);
    unsigned short* hbuf = (unsigned short*)(ws + OFF_H);

    hipMemsetAsync(counts, 0, Ee * sizeof(int), stream);
    hipMemsetAsync(d_out, 0, (size_t)out_size * sizeof(float), stream);

    router_kernel<<<Tt / 8, 256, 0, stream>>>(x, Wg, bg, counts, tokList, gateList);
    prefix_kernel<<<1, 32, 0, stream>>>(counts, offsets);
    // W1 [E][D][F] -> W1b [E][F][D] bf16 ; W2 [E][F][D] -> W2b [E][D][F] bf16
    transpose_cvt_kernel<<<dim3(Ff / 32, Dd / 32, Ee), dim3(32, 8), 0, stream>>>(W1, W1b, Dd, Ff);
    transpose_cvt_kernel<<<dim3(Dd / 32, Ff / 32, Ee), dim3(32, 8), 0, stream>>>(W2, W2b, Ff, Dd);

    gemm1_kernel<<<dim3(Tt / BM, Ff / BN, Ee), 256, 0, stream>>>(x, W1b, b1, counts, offsets, tokList, hbuf);
    gemm2_kernel<<<dim3(Tt / BM, Dd / BN, Ee), 256, 0, stream>>>(hbuf, W2b, b2, counts, offsets, tokList, gateList, out);
}
